// DozerAttention_19653770346745
// MI455X (gfx1250) — matrix operand order repaired and checked by reading, not compiled, not run on hardware
//
#include <hip/hip_runtime.h>

typedef _Float16 v16h __attribute__((ext_vector_type(16)));
typedef _Float16 v8h  __attribute__((ext_vector_type(8)));
typedef float    v8f  __attribute__((ext_vector_type(8)));
typedef float    v4f  __attribute__((ext_vector_type(4)));

#define B_   2
#define L_   2048
#define H_   8
#define D_   64
#define BQ   128          // query rows per block (8 waves x 16)
#define TK   32           // keys per tile
#define VSTR 40           // halves per transposed-V row (64B-friendly pad)
#define LOG2E 1.44269504088896340736f
#define NEGINF (-1e30f)

// Flash-attention, DozerAttention semantics:
//   score = (q.k)*scale if |i-j|<=16 or (i-j)%65==0 else 0, for j<=i; -inf for j>i
//   softmax over full row, then A @ V.
__global__ __launch_bounds__(256, 1)
void dozer_attn_fwd(const float* __restrict__ Qg,
                    const float* __restrict__ Kg,
                    const float* __restrict__ Vg,
                    float* __restrict__ Og)
{
    __shared__ _Float16 VT[D_][VSTR];      // V tile transposed: [d][k], f16
    __shared__ _Float16 Pl[8][16][TK];     // per-wave softmax-prob tile [m][k]

    const int tid  = threadIdx.x;
    const int w    = tid >> 5;
    const int lane = tid & 31;
    const int hl   = lane >> 4;            // lane half (WMMA layout select)
    const int c    = lane & 15;

    const int qt = blockIdx.x % (L_ / BQ);
    const int bh = blockIdx.x / (L_ / BQ);
    const int h  = bh % H_;
    const int b  = bh / H_;

    const int i0 = qt * BQ + w * 16;       // first query row of this wave
    const float scale = 0.125f;            // 1/sqrt(64)

    // ---- Q tile in WMMA A layout (16x32 halves per d-chunk), scale folded in ----
    // A layout: lane%16 = row M; lane<16 holds K={0..7,16..23}, lane>=16 K={8..15,24..31}
    const float* qp = Qg + (((size_t)b * L_ + (i0 + c)) * H_ + h) * D_;
    const int b0 = hl * 8;
    v16h qa0, qa1;
    {
        v4f q00 = *(const v4f*)(qp + b0),      q01 = *(const v4f*)(qp + b0 + 4);
        v4f q10 = *(const v4f*)(qp + b0 + 16), q11 = *(const v4f*)(qp + b0 + 20);
        v4f q20 = *(const v4f*)(qp + b0 + 32), q21 = *(const v4f*)(qp + b0 + 36);
        v4f q30 = *(const v4f*)(qp + b0 + 48), q31 = *(const v4f*)(qp + b0 + 52);
#pragma unroll
        for (int e = 0; e < 4; ++e) {
            qa0[e]      = (_Float16)(q00[e] * scale);
            qa0[e + 4]  = (_Float16)(q01[e] * scale);
            qa0[e + 8]  = (_Float16)(q10[e] * scale);
            qa0[e + 12] = (_Float16)(q11[e] * scale);
            qa1[e]      = (_Float16)(q20[e] * scale);
            qa1[e + 4]  = (_Float16)(q21[e] * scale);
            qa1[e + 8]  = (_Float16)(q30[e] * scale);
            qa1[e + 12] = (_Float16)(q31[e] * scale);
        }
    }

    v8f acc[4];
    const v8f vzero = {};
#pragma unroll
    for (int k4 = 0; k4 < 4; ++k4) acc[k4] = vzero;
    float m_run[8], l_run[8];
#pragma unroll
    for (int r = 0; r < 8; ++r) { m_run[r] = NEGINF; l_run[r] = 0.0f; }

    const int ktEnd = qt * 4 + 4;          // last key tile any wave in block needs
    for (int kt = 0; kt < ktEnd; ++kt) {
        const int j0 = kt * TK;
        __syncthreads();
        // ---- cooperative stage: V tile -> LDS transposed [d][k] as f16 ----
        {
            const int kk = tid >> 3;              // key 0..31
            const int db = (tid & 7) * 8;         // d base
            const float* vp = Vg + (((size_t)b * L_ + (j0 + kk)) * H_ + h) * D_ + db;
            v4f v0 = *(const v4f*)vp;
            v4f v1 = *(const v4f*)(vp + 4);
            if (kt + 1 < ktEnd) __builtin_prefetch(vp + TK * H_ * D_, 0, 1);
#pragma unroll
            for (int e = 0; e < 4; ++e) {
                VT[db + e][kk]     = (_Float16)v0[e];
                VT[db + 4 + e][kk] = (_Float16)v1[e];
            }
        }
        __syncthreads();

        if (j0 > i0 + 15) continue;        // all of this wave's rows precede tile

        // ---- S = Q K^T : two 16-key chunks, K-dim = D split in two 32-wide WMMAs ----
        v8f s[2];
#pragma unroll
        for (int nc = 0; nc < 2; ++nc) {
            // B layout: lane%16 = key column, lane<16 holds d={0..7,16..23}, lane>=16 d={8..15,24..31}
            const int kr = j0 + nc * 16 + c;
            const float* kp = Kg + (((size_t)b * L_ + kr) * H_ + h) * D_ + hl * 8;
            v16h kb0, kb1;
            v4f a0 = *(const v4f*)kp,        a1 = *(const v4f*)(kp + 4);
            v4f a2 = *(const v4f*)(kp + 16),  a3 = *(const v4f*)(kp + 20);
            v4f c0 = *(const v4f*)(kp + 32), c1 = *(const v4f*)(kp + 36);
            v4f c2 = *(const v4f*)(kp + 48), c3 = *(const v4f*)(kp + 52);
#pragma unroll
            for (int e = 0; e < 4; ++e) {
                kb0[e] = (_Float16)a0[e]; kb0[e+4] = (_Float16)a1[e];
                kb0[e+8] = (_Float16)a2[e]; kb0[e+12] = (_Float16)a3[e];
                kb1[e] = (_Float16)c0[e]; kb1[e+4] = (_Float16)c1[e];
                kb1[e+8] = (_Float16)c2[e]; kb1[e+12] = (_Float16)c3[e];
            }
            v8f t = __builtin_amdgcn_wmma_f32_16x16x32_f16(
                        false, qa0, false, kb0, (short)0, vzero, false, false);
            s[nc] = __builtin_amdgcn_wmma_f32_16x16x32_f16(
                        false, qa1, false, kb1, (short)0, t, false, false);
        }

        // ---- mask + online softmax (C layout: row m=r+8*hl, col n=c) ----
        float alpha[8];
#pragma unroll
        for (int r = 0; r < 8; ++r) {
            const int i  = i0 + r + 8 * hl;
            const int d0 = i - (j0 + c);
            const int d1 = i - (j0 + 16 + c);
            float x0 = (d0 < 0) ? NEGINF
                     : (((d0 <= 16) || (d0 % 65 == 0)) ? s[0][r] : 0.0f);
            float x1 = (d1 < 0) ? NEGINF
                     : (((d1 <= 16) || (d1 % 65 == 0)) ? s[1][r] : 0.0f);
            float lm = fmaxf(x0, x1);
#pragma unroll
            for (int mk = 1; mk < 16; mk <<= 1)
                lm = fmaxf(lm, __shfl_xor(lm, mk, 32));
            const float mnew = fmaxf(m_run[r], lm);
            const float al = exp2f((m_run[r] - mnew) * LOG2E);
            const float p0 = exp2f((x0 - mnew) * LOG2E);
            const float p1 = exp2f((x1 - mnew) * LOG2E);
            float rs = p0 + p1;
#pragma unroll
            for (int mk = 1; mk < 16; mk <<= 1)
                rs += __shfl_xor(rs, mk, 32);
            l_run[r] = l_run[r] * al + rs;
            m_run[r] = mnew;
            alpha[r] = al;
            const int mr = r + 8 * hl;
            Pl[w][mr][c]      = (_Float16)p0;
            Pl[w][mr][16 + c] = (_Float16)p1;
        }

        // ---- rescale running accumulator ----
#pragma unroll
        for (int k4 = 0; k4 < 4; ++k4)
#pragma unroll
            for (int r = 0; r < 8; ++r) acc[k4][r] *= alpha[r];

        // ---- P in A layout via per-wave LDS (DS ops are in-order per wave) ----
        v16h pa;
        {
            const v8h lo = *(const v8h*)&Pl[w][c][hl * 16];
            const v8h hi = *(const v8h*)&Pl[w][c][hl * 16 + 8];
#pragma unroll
            for (int e = 0; e < 8; ++e) { pa[e] = lo[e]; pa[e + 8] = hi[e]; }
        }

        // ---- O += P * V (B operand contiguous from transposed-V LDS) ----
#pragma unroll
        for (int k4 = 0; k4 < 4; ++k4) {
            const int dcol = k4 * 16 + c;
            const v8h vlo = *(const v8h*)&VT[dcol][hl * 16];
            const v8h vhi = *(const v8h*)&VT[dcol][hl * 16 + 8];
            v16h vb;
#pragma unroll
            for (int e = 0; e < 8; ++e) { vb[e] = vlo[e]; vb[e + 8] = vhi[e]; }
            acc[k4] = __builtin_amdgcn_wmma_f32_16x16x32_f16(
                          false, pa, false, vb, (short)0, acc[k4], false, false);
        }
    }

    // ---- normalize and write out [B,L,H,D] fp32 ----
#pragma unroll
    for (int k4 = 0; k4 < 4; ++k4) {
#pragma unroll
        for (int r = 0; r < 8; ++r) {
            const int i = i0 + r + 8 * hl;
            const int d = k4 * 16 + c;
            Og[(((size_t)b * L_ + i) * H_ + h) * D_ + d] = acc[k4][r] / l_run[r];
        }
    }
}

extern "C" void kernel_launch(void* const* d_in, const int* in_sizes, int n_in,
                              void* d_out, int out_size, void* d_ws, size_t ws_size,
                              hipStream_t stream) {
    (void)in_sizes; (void)n_in; (void)d_ws; (void)ws_size; (void)out_size;
    const float* Q = (const float*)d_in[0];
    const float* K = (const float*)d_in[1];
    const float* V = (const float*)d_in[2];
    // d_in[3] = boolean causal mask: implemented analytically in-kernel.
    float* O = (float*)d_out;
    dim3 grid(B_ * H_ * (L_ / BQ));   // 256 blocks
    dim3 block(256);                  // 8 waves (wave32)
    hipLaunchKernelGGL(dozer_attn_fwd, grid, block, 0, stream, Q, K, V, O);
}
